// PatchClusterBlock_20512763805898
// MI455X (gfx1250) — matrix operand order repaired and checked by reading, not compiled, not run on hardware
//
#include <hip/hip_runtime.h>
#include <math.h>

// Problem constants (from reference): B=16, N=1024, C=768, clusters=256, k=5
#define Bq  16
#define Nq  1024
#define Cq  768
#define CL  256

typedef __attribute__((ext_vector_type(16))) __bf16 v16bf;
typedef __attribute__((ext_vector_type(8)))  float  v8f;
typedef unsigned short u16;
typedef unsigned int   u32;

// ---- bf16 helpers (manual RNE, storage as ushort; no __bf16 arithmetic) ----
__device__ __forceinline__ u16 f2bf_rne(float f) {
    u32 u = __float_as_uint(f);
    u32 r = u + 0x7FFFu + ((u >> 16) & 1u);
    return (u16)(r >> 16);
}
__device__ __forceinline__ float bf2f(u16 h) {
    return __uint_as_float(((u32)h) << 16);
}

// ============================================================================
// K0: per-row squared norm + fp32 -> bf16 hi/lo split (one wave32 per row)
// ============================================================================
__global__ void k_prep(const float* __restrict__ x, u16* __restrict__ xhi,
                       u16* __restrict__ xlo, float* __restrict__ x2) {
    int row  = blockIdx.x * 8 + (threadIdx.x >> 5);   // b*N + n
    int lane = threadIdx.x & 31;
    const float* xr = x + (size_t)row * Cq;
    u16* hr = xhi + (size_t)row * Cq;
    u16* lr = xlo + (size_t)row * Cq;
    float s = 0.0f;
    for (int c = lane; c < Cq; c += 32) {
        float v = xr[c];
        s += v * v;
        u16 h = f2bf_rne(v);
        hr[c] = h;
        lr[c] = f2bf_rne(v - bf2f(h));
    }
    #pragma unroll
    for (int off = 16; off; off >>= 1) s += __shfl_xor(s, off, 32);
    if (lane == 0) x2[row] = s;
}

// ============================================================================
// K1: dist[b][i][j] = sqrt(max(|xi|^2+|xj|^2-2 xi.xj,0))/sqrt(C)
// One wave computes a 32x64 block (2 A-row tiles x 4 B-col tiles, 8 f32
// accumulators). Gram via v_wmma_f32_16x16x32_bf16 with hi/lo split
// (hi*hi + hi*lo + lo*hi) for ~fp32 accuracy: 24 WMMAs per K=32 step.
// Symmetry exploited: blocks entirely below the diagonal are skipped; each
// 16x16 subtile on/above the diagonal is stored (and mirrored if strictly
// above). All branches are wave-uniform, so EXEC stays all-ones for WMMA.
// ============================================================================
__global__ void __launch_bounds__(32)
k_dist(const u16* __restrict__ xhi, const u16* __restrict__ xlo,
       const float* __restrict__ x2, float* __restrict__ dist) {
    const int b  = blockIdx.z;
    const int ti = blockIdx.x;                 // 32-row block: rows ti*32..+31
    const int tj = blockIdx.y;                 // 64-col block: cols tj*64..+63
    if (tj * 4 + 3 < ti * 2) return;           // entire block strictly below diag

    const int lane = threadIdx.x;
    const int half = lane >> 4, l16 = lane & 15;
    const size_t base = (size_t)b * Nq * Cq;

    const u16* A0h = xhi + base + (size_t)(ti * 32 + l16) * Cq;
    const u16* A0l = xlo + base + (size_t)(ti * 32 + l16) * Cq;
    const u16* A1h = A0h + (size_t)16 * Cq;
    const u16* A1l = A0l + (size_t)16 * Cq;

    union U { u16 u[16]; v16bf v; };
    v8f acc[2][4];
    #pragma unroll
    for (int m = 0; m < 2; ++m)
        #pragma unroll
        for (int j = 0; j < 4; ++j)
            acc[m][j] = (v8f){0.f, 0.f, 0.f, 0.f, 0.f, 0.f, 0.f, 0.f};

    for (int k0 = 0; k0 < Cq; k0 += 32) {
        // A fragments (16-bit A 16x32 layout): two contiguous 8-elem K chunks
        const int ka = k0 + half * 8;
        __builtin_prefetch(A0h + ka + 32, 0, 1);
        __builtin_prefetch(A1h + ka + 32, 0, 1);
        U a0h, a0l, a1h, a1l;
        #pragma unroll
        for (int e = 0; e < 8; ++e) {
            a0h.u[e] = A0h[ka + e];  a0h.u[e + 8] = A0h[ka + 16 + e];
            a0l.u[e] = A0l[ka + e];  a0l.u[e + 8] = A0l[ka + 16 + e];
            a1h.u[e] = A1h[ka + e];  a1h.u[e + 8] = A1h[ka + 16 + e];
            a1l.u[e] = A1l[ka + e];  a1l.u[e + 8] = A1l[ka + 16 + e];
        }
        // 4 B-col tiles reuse the A fragments
        #pragma unroll
        for (int j = 0; j < 4; ++j) {
            const size_t coff = base + (size_t)(tj * 64 + j * 16 + l16) * Cq
                              + (size_t)(k0 + half * 8);
            const u16* Bph = xhi + coff;
            const u16* Bpl = xlo + coff;
            U bh, bl;
            #pragma unroll
            for (int e = 0; e < 16; ++e) { bh.u[e] = Bph[e < 8 ? e : e + 8]; bl.u[e] = Bpl[e < 8 ? e : e + 8]; }

            acc[0][j] = __builtin_amdgcn_wmma_f32_16x16x32_bf16(
                false, a0h.v, false, bh.v, (short)0, acc[0][j], false, false);
            acc[0][j] = __builtin_amdgcn_wmma_f32_16x16x32_bf16(
                false, a0h.v, false, bl.v, (short)0, acc[0][j], false, false);
            acc[0][j] = __builtin_amdgcn_wmma_f32_16x16x32_bf16(
                false, a0l.v, false, bh.v, (short)0, acc[0][j], false, false);

            acc[1][j] = __builtin_amdgcn_wmma_f32_16x16x32_bf16(
                false, a1h.v, false, bh.v, (short)0, acc[1][j], false, false);
            acc[1][j] = __builtin_amdgcn_wmma_f32_16x16x32_bf16(
                false, a1h.v, false, bl.v, (short)0, acc[1][j], false, false);
            acc[1][j] = __builtin_amdgcn_wmma_f32_16x16x32_bf16(
                false, a1l.v, false, bh.v, (short)0, acc[1][j], false, false);
        }
    }

    const float inv = 0.03608439182435161f;    // 1/sqrt(768)
    float* D = dist + (size_t)b * Nq * Nq;
    #pragma unroll
    for (int m = 0; m < 2; ++m) {
        #pragma unroll
        for (int j = 0; j < 4; ++j) {
            const int I = ti * 2 + m;          // 16-row subtile index
            const int J = tj * 4 + j;          // 16-col subtile index
            if (J < I) continue;               // covered by mirror of another block
            const int   jc = J * 16 + l16;
            const float xj = x2[b * Nq + jc];
            #pragma unroll
            for (int v = 0; v < 8; ++v) {      // C/D layout: row = v + 8*half
                int   i  = I * 16 + v + half * 8;
                float d2 = x2[b * Nq + i] + xj - 2.0f * acc[m][j][v];
                float dv = sqrtf(fmaxf(d2, 0.0f)) * inv;
                D[(size_t)i * Nq + jc] = dv;
                if (J > I) D[(size_t)jc * Nq + i] = dv;   // mirrored store
            }
        }
    }
}

// ============================================================================
// K2: density = exp(-mean(k=5 smallest dist^2)) per token (one wave per row)
// ============================================================================
__device__ __forceinline__ void ins5(float v, float& t0, float& t1, float& t2,
                                     float& t3, float& t4) {
    if (v < t4) {
        t4 = v;
        if (t4 < t3) { float w = t3; t3 = t4; t4 = w; }
        if (t3 < t2) { float w = t2; t2 = t3; t3 = w; }
        if (t2 < t1) { float w = t1; t1 = t2; t2 = w; }
        if (t1 < t0) { float w = t0; t0 = t1; t1 = w; }
    }
}

__global__ void k_density(const float* __restrict__ dist, float* __restrict__ density) {
    int row  = blockIdx.x * 8 + (threadIdx.x >> 5);
    int lane = threadIdx.x & 31;
    const float* dr = dist + (size_t)row * Nq;
    float t0 = 3.4e38f, t1 = 3.4e38f, t2 = 3.4e38f, t3 = 3.4e38f, t4 = 3.4e38f;
    for (int j = lane; j < Nq; j += 32) ins5(dr[j], t0, t1, t2, t3, t4);
    #pragma unroll
    for (int off = 16; off; off >>= 1) {
        float o0 = __shfl_xor(t0, off, 32), o1 = __shfl_xor(t1, off, 32);
        float o2 = __shfl_xor(t2, off, 32), o3 = __shfl_xor(t3, off, 32);
        float o4 = __shfl_xor(t4, off, 32);
        ins5(o0, t0, t1, t2, t3, t4); ins5(o1, t0, t1, t2, t3, t4);
        ins5(o2, t0, t1, t2, t3, t4); ins5(o3, t0, t1, t2, t3, t4);
        ins5(o4, t0, t1, t2, t3, t4);
    }
    if (lane == 0)
        density[row] = expf(-(t0*t0 + t1*t1 + t2*t2 + t3*t3 + t4*t4) * 0.2f);
}

// ============================================================================
// K3: per-batch max distance (one block per batch, deterministic LDS reduce)
// ============================================================================
__global__ void k_dmax(const float* __restrict__ dist, float* __restrict__ dmax) {
    int b = blockIdx.x, t = threadIdx.x;
    const float* d = dist + (size_t)b * Nq * Nq;
    float m = 0.0f;
    for (int i = t; i < Nq * Nq; i += 256) m = fmaxf(m, d[i]);
    __shared__ float sm[256];
    sm[t] = m; __syncthreads();
    for (int s = 128; s; s >>= 1) {
        if (t < s) sm[t] = fmaxf(sm[t], sm[t + s]);
        __syncthreads();
    }
    if (t == 0) dmax[b] = sm[0];
}

// ============================================================================
// K4: score = density * min(dmax, min over higher-density j of dist[i][j])
// ============================================================================
__global__ void k_score(const float* __restrict__ dist, const float* __restrict__ density,
                        const float* __restrict__ dmax, float* __restrict__ score) {
    int row  = blockIdx.x * 8 + (threadIdx.x >> 5);
    int b    = row >> 10;
    int lane = threadIdx.x & 31;
    const float* dr   = dist + (size_t)row * Nq;
    const float* dens = density + (size_t)b * Nq;
    float di    = density[row];
    float local = dmax[b];
    for (int j = lane; j < Nq; j += 32)
        if (dens[j] > di) local = fminf(local, dr[j]);
    #pragma unroll
    for (int off = 16; off; off >>= 1) local = fminf(local, __shfl_xor(local, off, 32));
    if (lane == 0) score[row] = local * di;
}

// ============================================================================
// K5: exact top-256 by rank counting (stable: ties -> lower index first,
// matching jax.lax.top_k). Writes index_down (int), center_rank inverse map,
// and the index_down portion of d_out as floats.
// ============================================================================
__global__ void k_select(const float* __restrict__ score, int* __restrict__ index_down,
                         int* __restrict__ center_rank, float* __restrict__ out_idx) {
    int b = blockIdx.x, t = threadIdx.x;
    __shared__ float s[Nq];
    const float* sr = score + (size_t)b * Nq;
    for (int i = t; i < Nq; i += 256) s[i] = sr[i];
    __syncthreads();
    for (int i = t; i < Nq; i += 256) {
        float mine = s[i];
        int rank = 0;
        for (int j = 0; j < Nq; ++j) {
            float oj = s[j];
            rank += (oj > mine) || (oj == mine && j < i);
        }
        if (rank < CL) {
            index_down[b * CL + rank] = i;
            out_idx[b * CL + rank]    = (float)i;   // output dtype is float
            center_rank[b * Nq + i]   = rank;
        } else {
            center_rank[b * Nq + i] = -1;
        }
    }
}

// ============================================================================
// K6: assign each token to nearest center (first-wins argmin, k ascending);
// centers forced to their own cluster id via center_rank.
// ============================================================================
__global__ void k_assign(const float* __restrict__ dist, const int* __restrict__ index_down,
                         const int* __restrict__ center_rank, int* __restrict__ idx_cluster) {
    int gid = blockIdx.x * 256 + threadIdx.x;   // b*N + n
    int b = gid >> 10, n = gid & 1023;
    int r = center_rank[gid];
    int best_k = 0;
    if (r >= 0) {
        best_k = r;
    } else {
        const float* db  = dist + (size_t)b * Nq * Nq;
        const int*   idd = index_down + b * CL;
        float best = 3.4e38f;
        for (int k = 0; k < CL; ++k) {
            float d = db[(size_t)idd[k] * Nq + n];   // coalesced across n
            if (d < best) { best = d; best_k = k; }
        }
    }
    idx_cluster[gid] = best_k;
}

// ============================================================================
// K7: deterministic merge: merged[b][k][c] = sum_{n in cluster k} x / (cnt+1e-6)
// One block per (cluster, batch); 256 threads cover 768 channels (3 each).
// ============================================================================
__global__ void k_merge(const float* __restrict__ x, const int* __restrict__ idx_cluster,
                        float* __restrict__ merged) {
    int k = blockIdx.x, b = blockIdx.y, t = threadIdx.x;
    const int*   ic = idx_cluster + b * Nq;
    const float* xb = x + (size_t)b * Nq * Cq;
    float s0 = 0.f, s1 = 0.f, s2 = 0.f;
    int cnt = 0;
    for (int n = 0; n < Nq; ++n) {
        if (ic[n] == k) {                     // broadcast scalar load
            const float* xr = xb + (size_t)n * Cq;
            s0 += xr[t]; s1 += xr[t + 256]; s2 += xr[t + 512];
            ++cnt;
        }
    }
    float inv = 1.0f / ((float)cnt + 1e-6f);
    float* m = merged + (size_t)(b * CL + k) * Cq;
    m[t] = s0 * inv; m[t + 256] = s1 * inv; m[t + 512] = s2 * inv;
}

// ============================================================================
extern "C" void kernel_launch(void* const* d_in, const int* in_sizes, int n_in,
                              void* d_out, int out_size, void* d_ws, size_t ws_size,
                              hipStream_t stream) {
    const float* x   = (const float*)d_in[0];
    float*       out = (float*)d_out;   // [B*CL] index_down (as float) ++ [B*CL*C] merged

    // ---- workspace carve-out (all 256B aligned) ----
    char*  base = (char*)d_ws;
    size_t o = 0;
    auto take = [&](size_t n) -> char* {
        char* p = base + o;
        o += (n + 255) & ~(size_t)255;
        return p;
    };
    u16*   xhi         = (u16*)  take((size_t)Bq * Nq * Cq * sizeof(u16));   // 24 MB
    u16*   xlo         = (u16*)  take((size_t)Bq * Nq * Cq * sizeof(u16));   // 24 MB
    float* x2          = (float*)take((size_t)Bq * Nq * sizeof(float));
    float* dist        = (float*)take((size_t)Bq * Nq * Nq * sizeof(float)); // 64 MB (L2-resident)
    float* density     = (float*)take((size_t)Bq * Nq * sizeof(float));
    float* dmax        = (float*)take((size_t)Bq * sizeof(float));
    float* score       = (float*)take((size_t)Bq * Nq * sizeof(float));
    int*   index_down  = (int*)  take((size_t)Bq * CL * sizeof(int));
    int*   center_rank = (int*)  take((size_t)Bq * Nq * sizeof(int));
    int*   idx_cluster = (int*)  take((size_t)Bq * Nq * sizeof(int));
    (void)ws_size; (void)in_sizes; (void)n_in; (void)out_size;

    // ---- pipeline (stream-ordered, deterministic, capture-safe) ----
    k_prep   <<<dim3(Bq * Nq / 8), 256, 0, stream>>>(x, xhi, xlo, x2);
    k_dist   <<<dim3(Nq / 32, Nq / 64, Bq), 32, 0, stream>>>(xhi, xlo, x2, dist);
    k_density<<<dim3(Bq * Nq / 8), 256, 0, stream>>>(dist, density);
    k_dmax   <<<dim3(Bq), 256, 0, stream>>>(dist, dmax);
    k_score  <<<dim3(Bq * Nq / 8), 256, 0, stream>>>(dist, density, dmax, score);
    k_select <<<dim3(Bq), 256, 0, stream>>>(score, index_down, center_rank, out);
    k_assign <<<dim3(Bq * Nq / 256), 256, 0, stream>>>(dist, index_down, center_rank, idx_cluster);
    k_merge  <<<dim3(CL, Bq), 256, 0, stream>>>(x, idx_cluster, out + (size_t)Bq * CL);
}
